// DeepSSMConditioned_7799660610128
// MI455X (gfx1250) — matrix operand order repaired and checked by reading, not compiled, not run on hardware
//
#include <hip/hip_runtime.h>
#include <hip/hip_bf16.h>
#include <math.h>

#define B_  256
#define T_  512
#define DY_ 32
#define DU_ 16
#define Z_  64
#define H_  256
#define HM_ 512

typedef __attribute__((ext_vector_type(16))) _Float16 v16h;
typedef __attribute__((ext_vector_type(8)))  float    v8f;

// ---------------------------------------------------------------------------
// WMMA helpers (CDNA5 v_wmma_f32_16x16x32_f16, wave32)
// A (16x32 f16) per-lane layout: lane L: M=L&15, half=(L>>4);
//   elems 0..15 -> K = kbase + half*16 + e
// B (32x16 f16) per-lane layout: lane L: N=L&15;
//   elems 0..15 -> K = kbase + (L>>4)*16 + e   (B[k][n] = W[n][k], W row-major [N][K])
// C/D (16x16 f32): VGPR r, lane L: M = r + 8*(L>>4), N = L&15
// ---------------------------------------------------------------------------
__device__ __forceinline__ v16h ldA(const _Float16* __restrict__ buf, int stride,
                                    int m, int khb, int kbase) {
  v16h a;
  const _Float16* p = buf + m * stride + kbase + 2 * khb;
#pragma unroll
  for (int e = 0; e < 8; ++e) a[e] = p[e];
#pragma unroll
  for (int e = 0; e < 8; ++e) a[8 + e] = p[8 + e];
  return a;
}

__device__ __forceinline__ v16h ldB(const _Float16* __restrict__ W, int K,
                                    int n, int kk) {
  v16h b;
  const _Float16* p = W + (size_t)n * K + kk;
#pragma unroll
  for (int e = 0; e < 16; ++e) b[e] = p[e];
  return b;
}

__device__ __forceinline__ v8f wmma16(v16h a, v16h b, v8f c) {
  return __builtin_amdgcn_wmma_f32_16x16x32_f16(false, a, false, b, (short)0, c,
                                                false, false);
}

__device__ __forceinline__ float sigm_f(float x) { return 1.f / (1.f + expf(-x)); }
__device__ __forceinline__ float silu_f(float x) { return x / (1.f + expf(-x)); }

// ---------------------------------------------------------------------------
// Weight conversion kernels (f32 -> f16, optional K padding with zeros)
// ---------------------------------------------------------------------------
__global__ void cvt_f32_f16(const float* __restrict__ s, _Float16* __restrict__ d, int n) {
  int i = blockIdx.x * blockDim.x + threadIdx.x;
  if (i < n) d[i] = (_Float16)s[i];
}

__global__ void cvt_pad_f16(const float* __restrict__ s, _Float16* __restrict__ d,
                            int rows, int ksrc, int kdst) {
  int i = blockIdx.x * blockDim.x + threadIdx.x;
  if (i < rows * kdst) {
    int r = i / kdst, k = i - r * kdst;
    d[i] = (_Float16)(k < ksrc ? s[r * ksrc + k] : 0.f);
  }
}

__global__ void zero2_kernel(float* out) { out[0] = 0.f; out[1] = 0.f; }

// ---------------------------------------------------------------------------
// Fused GRU scan + posterior projection + reparameterized sample.
// grid = B/16 workgroups, 256 threads (8 waves). Each WG owns 16 batch rows.
// Hidden state: f32 master copy in registers (thread tid owns column tid of
// all 16 rows), f16 mirror in LDS as WMMA A operand.
// ---------------------------------------------------------------------------
__global__ void __launch_bounds__(256)
gru_scan_kernel(const float* __restrict__ y, const float* __restrict__ u,
                const float* __restrict__ eps,
                const _Float16* __restrict__ wih,  // [768][64]  (K padded 48->64)
                const _Float16* __restrict__ whh,  // [768][256]
                const _Float16* __restrict__ wqm,  // [128][256]
                const float* __restrict__ bih, const float* __restrict__ bhh,
                const float* __restrict__ bq,
                float* __restrict__ mu_q, float* __restrict__ lv_q,
                _Float16* __restrict__ z_h) {
  __shared__ __align__(16) _Float16 xs[16 * 64];    //  2 KB  x_t = [y|u|0] f16
  __shared__ __align__(16) _Float16 hf[16 * H_];    //  8 KB  h (f16 mirror)
  __shared__ __align__(16) _Float16 rz[16 * 512];   // 16 KB  sigmoid(r),sigmoid(z)
  __shared__ __align__(16) float    gn[16 * 512];   // 32 KB  n-gate i/h preacts; reused as q

  const int tid  = threadIdx.x;
  const int wave = tid >> 5, lane = tid & 31;
  const int ml = lane & 15, hh = lane >> 4, khb = hh * 8;
  const int b0 = blockIdx.x * 16;

  float hreg[16];
#pragma unroll
  for (int j = 0; j < 16; ++j) hreg[j] = 0.f;
  for (int i = tid; i < 16 * H_; i += 256) hf[i] = (_Float16)0.f;
  __syncthreads();

  for (int t = 0; t < T_; ++t) {
    // ---- stage x_t into LDS (f16, K padded to 64) ----
    for (int i = tid; i < 16 * 64; i += 256) {
      int mm = i >> 6, k = i & 63;
      float v = 0.f;
      if (k < DY_)            v = y[((size_t)(b0 + mm) * T_ + t) * DY_ + k];
      else if (k < DY_ + DU_) v = u[((size_t)(b0 + mm) * T_ + t) * DU_ + (k - DY_)];
      xs[i] = (_Float16)v;
    }
    __syncthreads();

    // ---- 64 virtual 16x16 gate tiles, interleaved across waves ----
    // vt  0..31: r,z cols 0..511   (x-part + h-part, both biases, sigmoid -> f16)
    // vt 32..47: n cols, input part (x only, b_ih)  -> gn[:, 0..255]
    // vt 48..63: n cols, hidden part (h only, b_hh) -> gn[:, 256..511]
#pragma unroll 1
    for (int tt = 0; tt < 8; ++tt) {
      int vt = tt * 8 + wave;
      v8f acc;
      if (vt < 32) {
        int n0 = vt * 16;
        float bias = bih[n0 + ml] + bhh[n0 + ml];
#pragma unroll
        for (int r = 0; r < 8; ++r) acc[r] = bias;
#pragma unroll
        for (int kb = 0; kb < 64; kb += 32)
          acc = wmma16(ldA(xs, 64, ml, khb, kb), ldB(wih, 64, n0 + ml, kb + hh * 16), acc);
#pragma unroll
        for (int kb = 0; kb < H_; kb += 32)
          acc = wmma16(ldA(hf, H_, ml, khb, kb), ldB(whh, H_, n0 + ml, kb + hh * 16), acc);
#pragma unroll
        for (int r = 0; r < 8; ++r)
          rz[(r + 8 * hh) * 512 + n0 + ml] = (_Float16)sigm_f(acc[r]);
      } else if (vt < 48) {
        int c = (vt - 32) * 16, n0 = 512 + c;
        float bias = bih[n0 + ml];
#pragma unroll
        for (int r = 0; r < 8; ++r) acc[r] = bias;
#pragma unroll
        for (int kb = 0; kb < 64; kb += 32)
          acc = wmma16(ldA(xs, 64, ml, khb, kb), ldB(wih, 64, n0 + ml, kb + hh * 16), acc);
#pragma unroll
        for (int r = 0; r < 8; ++r) gn[(r + 8 * hh) * 512 + c + ml] = acc[r];
      } else {
        int c = (vt - 48) * 16, n0 = 512 + c;
        float bias = bhh[n0 + ml];
#pragma unroll
        for (int r = 0; r < 8; ++r) acc[r] = bias;
#pragma unroll
        for (int kb = 0; kb < H_; kb += 32)
          acc = wmma16(ldA(hf, H_, ml, khb, kb), ldB(whh, H_, n0 + ml, kb + hh * 16), acc);
#pragma unroll
        for (int r = 0; r < 8; ++r) gn[(r + 8 * hh) * 512 + 256 + c + ml] = acc[r];
      }
    }
    __syncthreads();

    // ---- elementwise GRU update: thread tid owns h[:, tid] ----
#pragma unroll
    for (int j = 0; j < 16; ++j) {
      float rg = (float)rz[j * 512 + tid];
      float zg = (float)rz[j * 512 + 256 + tid];
      float nn = tanhf(gn[j * 512 + tid] + rg * gn[j * 512 + 256 + tid]);
      float hv = (1.f - zg) * nn + zg * hreg[j];
      hreg[j] = hv;
      hf[j * H_ + tid] = (_Float16)hv;
    }
    __syncthreads();

    // ---- posterior projection: q = h @ wq^T + bq  (128 cols, 1 tile/wave) ----
    {
      int n0 = wave * 16;
      v8f acc;
      float bias = bq[n0 + ml];
#pragma unroll
      for (int r = 0; r < 8; ++r) acc[r] = bias;
#pragma unroll
      for (int kb = 0; kb < H_; kb += 32)
        acc = wmma16(ldA(hf, H_, ml, khb, kb), ldB(wqm, H_, n0 + ml, kb + hh * 16), acc);
#pragma unroll
      for (int r = 0; r < 8; ++r) gn[(r + 8 * hh) * 128 + n0 + ml] = acc[r];
    }
    __syncthreads();

    // ---- reparameterized sample + outputs ----
    for (int i = tid; i < 16 * Z_; i += 256) {
      int mm = i >> 6, zc = i & 63;
      float mu = gn[mm * 128 + zc];
      float lv = fminf(6.f, fmaxf(-12.f, gn[mm * 128 + 64 + zc]));
      size_t idx = ((size_t)(b0 + mm) * T_ + t) * Z_ + zc;
      float zv = mu + eps[idx] * expf(0.5f * lv);
      mu_q[idx] = mu;
      lv_q[idx] = lv;
      z_h[idx] = (_Float16)zv;
    }
    __syncthreads();
  }
}

// ---------------------------------------------------------------------------
// Emission MLP (z -> 512 -> 512 -> 32) + NLL partial reduction.
// grid = B*T/16 blocks, 256 threads; one 16-row tile per block.
// ---------------------------------------------------------------------------
__global__ void __launch_bounds__(256)
emission_nll_kernel(const _Float16* __restrict__ z_h, const float* __restrict__ y,
                    const _Float16* __restrict__ ew1, const float* __restrict__ eb1,
                    const _Float16* __restrict__ ew2, const float* __restrict__ eb2,
                    const _Float16* __restrict__ ew3, const float* __restrict__ eb3,
                    const float* __restrict__ lsy, float* __restrict__ out) {
  __shared__ __align__(16) _Float16 a0[16 * 64];
  __shared__ __align__(16) _Float16 h1[16 * HM_];
  __shared__ __align__(16) _Float16 h2[16 * HM_];
  __shared__ float sq[16 * DY_];
  __shared__ float red[16];

  const int tid  = threadIdx.x;
  const int wave = tid >> 5, lane = tid & 31;
  const int ml = lane & 15, hh = lane >> 4, khb = hh * 8;
  const size_t row0 = (size_t)blockIdx.x * 16;

  for (int i = tid; i < 16 * 64; i += 256) a0[i] = z_h[row0 * 64 + i];
  __syncthreads();

#pragma unroll 1
  for (int tt = 0; tt < 4; ++tt) {  // stage1: K=64 -> 512, silu
    int n0 = (wave * 4 + tt) * 16;
    v8f acc;
    float bias = eb1[n0 + ml];
#pragma unroll
    for (int r = 0; r < 8; ++r) acc[r] = bias;
#pragma unroll
    for (int kb = 0; kb < 64; kb += 32)
      acc = wmma16(ldA(a0, 64, ml, khb, kb), ldB(ew1, 64, n0 + ml, kb + hh * 16), acc);
#pragma unroll
    for (int r = 0; r < 8; ++r) h1[(r + 8 * hh) * HM_ + n0 + ml] = (_Float16)silu_f(acc[r]);
  }
  __syncthreads();

#pragma unroll 1
  for (int tt = 0; tt < 4; ++tt) {  // stage2: K=512 -> 512, silu
    int n0 = (wave * 4 + tt) * 16;
    v8f acc;
    float bias = eb2[n0 + ml];
#pragma unroll
    for (int r = 0; r < 8; ++r) acc[r] = bias;
#pragma unroll 1
    for (int kb = 0; kb < HM_; kb += 32)
      acc = wmma16(ldA(h1, HM_, ml, khb, kb), ldB(ew2, HM_, n0 + ml, kb + hh * 16), acc);
#pragma unroll
    for (int r = 0; r < 8; ++r) h2[(r + 8 * hh) * HM_ + n0 + ml] = (_Float16)silu_f(acc[r]);
  }
  __syncthreads();

  float cls = fminf(3.f, fmaxf(-6.f, lsy[0]));
  float inv_sig2 = expf(-2.f * cls);
  if (wave < 2) {  // stage3: K=512 -> 32 (wave-uniform branch, EXEC all-1 inside)
    int n0 = wave * 16;
    v8f acc;
    float bias = eb3[n0 + ml];
#pragma unroll
    for (int r = 0; r < 8; ++r) acc[r] = bias;
#pragma unroll 1
    for (int kb = 0; kb < HM_; kb += 32)
      acc = wmma16(ldA(h2, HM_, ml, khb, kb), ldB(ew3, HM_, n0 + ml, kb + hh * 16), acc);
#pragma unroll
    for (int r = 0; r < 8; ++r) {
      int mrow = r + 8 * hh;
      float d = y[(row0 + mrow) * DY_ + n0 + ml] - acc[r];
      sq[mrow * DY_ + n0 + ml] = 0.5f * d * d * inv_sig2;
    }
  }
  __syncthreads();
  if (tid < 16) {
    float s = (float)DY_ * cls;
    for (int n = 0; n < DY_; ++n) s += sq[tid * DY_ + n];
    red[tid] = s;
  }
  __syncthreads();
  if (tid == 0) {
    float tot = 0.f;
    for (int i = 0; i < 16; ++i) tot += red[i];
    atomicAdd(out, tot * (1.f / (float)B_));
  }
}

// ---------------------------------------------------------------------------
// Transition MLP ([z_{t-1}|u_t] -> 512 -> 512 -> 128) + KL (t>=1) partial sum.
// grid = B * 32 blocks; block handles (b, tp0..tp0+15), tp < T-1 valid.
// ---------------------------------------------------------------------------
__global__ void __launch_bounds__(256)
trans_kl_kernel(const _Float16* __restrict__ z_h, const float* __restrict__ u,
                const float* __restrict__ mu_q, const float* __restrict__ lv_q,
                const _Float16* __restrict__ tw1, const float* __restrict__ tb1,
                const _Float16* __restrict__ tw2, const float* __restrict__ tb2,
                const _Float16* __restrict__ tw3, const float* __restrict__ tb3,
                float* __restrict__ out) {
  __shared__ __align__(16) _Float16 a0[16 * 96];
  __shared__ __align__(16) _Float16 h1[16 * HM_];
  __shared__ __align__(16) _Float16 h2[16 * HM_];
  __shared__ float qb[16 * 128];
  __shared__ float sq[16 * Z_];
  __shared__ float red[16];

  const int tid  = threadIdx.x;
  const int wave = tid >> 5, lane = tid & 31;
  const int ml = lane & 15, hh = lane >> 4, khb = hh * 8;
  const int b   = blockIdx.x >> 5;
  const int tp0 = (blockIdx.x & 31) * 16;

  for (int i = tid; i < 16 * 96; i += 256) {  // stage input [z|u|0], K 80->96
    int mm = i / 96, k = i - mm * 96;
    int tp = tp0 + mm;
    float v = 0.f;
    if (tp < T_ - 1) {
      if (k < Z_)            v = (float)z_h[((size_t)b * T_ + tp) * Z_ + k];
      else if (k < Z_ + DU_) v = u[((size_t)b * T_ + tp + 1) * DU_ + (k - Z_)];
    }
    a0[i] = (_Float16)v;
  }
  __syncthreads();

#pragma unroll 1
  for (int tt = 0; tt < 4; ++tt) {  // stage1: K=96 -> 512
    int n0 = (wave * 4 + tt) * 16;
    v8f acc;
    float bias = tb1[n0 + ml];
#pragma unroll
    for (int r = 0; r < 8; ++r) acc[r] = bias;
#pragma unroll
    for (int kb = 0; kb < 96; kb += 32)
      acc = wmma16(ldA(a0, 96, ml, khb, kb), ldB(tw1, 96, n0 + ml, kb + hh * 16), acc);
#pragma unroll
    for (int r = 0; r < 8; ++r) h1[(r + 8 * hh) * HM_ + n0 + ml] = (_Float16)silu_f(acc[r]);
  }
  __syncthreads();

#pragma unroll 1
  for (int tt = 0; tt < 4; ++tt) {  // stage2: K=512 -> 512
    int n0 = (wave * 4 + tt) * 16;
    v8f acc;
    float bias = tb2[n0 + ml];
#pragma unroll
    for (int r = 0; r < 8; ++r) acc[r] = bias;
#pragma unroll 1
    for (int kb = 0; kb < HM_; kb += 32)
      acc = wmma16(ldA(h1, HM_, ml, khb, kb), ldB(tw2, HM_, n0 + ml, kb + hh * 16), acc);
#pragma unroll
    for (int r = 0; r < 8; ++r) h2[(r + 8 * hh) * HM_ + n0 + ml] = (_Float16)silu_f(acc[r]);
  }
  __syncthreads();

  {  // stage3: K=512 -> 128 (1 tile/wave)
    int n0 = wave * 16;
    v8f acc;
    float bias = tb3[n0 + ml];
#pragma unroll
    for (int r = 0; r < 8; ++r) acc[r] = bias;
#pragma unroll 1
    for (int kb = 0; kb < HM_; kb += 32)
      acc = wmma16(ldA(h2, HM_, ml, khb, kb), ldB(tw3, HM_, n0 + ml, kb + hh * 16), acc);
#pragma unroll
    for (int r = 0; r < 8; ++r) qb[(r + 8 * hh) * 128 + n0 + ml] = acc[r];
  }
  __syncthreads();

  for (int i = tid; i < 16 * Z_; i += 256) {  // per-element KL
    int mm = i >> 6, zc = i & 63;
    int tp = tp0 + mm;
    float v = 0.f;
    if (tp < T_ - 1) {
      size_t idx = ((size_t)b * T_ + tp + 1) * Z_ + zc;
      float mq = mu_q[idx], lq = lv_q[idx];
      float mp = qb[mm * 128 + zc];
      float lp = fminf(6.f, fmaxf(-12.f, qb[mm * 128 + 64 + zc]));
      float dm = mq - mp;
      v = 0.5f * (lp - lq + (expf(lq) + dm * dm) * expf(-lp) - 1.f);
    }
    sq[i] = v;
  }
  __syncthreads();
  if (tid < 16) {
    float s = 0.f;
    for (int zc = 0; zc < Z_; ++zc) s += sq[tid * Z_ + zc];
    red[tid] = ((tp0 + tid) < T_ - 1) ? fmaxf(s, 0.5f) : 0.f;  // free bits
  }
  __syncthreads();
  if (tid == 0) {
    float tot = 0.f;
    for (int i = 0; i < 16; ++i) tot += red[i];
    atomicAdd(out + 1, tot * (1.f / (float)B_));
  }
}

// ---------------------------------------------------------------------------
// t=0 KL against N(0, I), free bits, add to out[1].
// ---------------------------------------------------------------------------
__global__ void kl0_kernel(const float* __restrict__ mu_q, const float* __restrict__ lv_q,
                           float* __restrict__ out) {
  __shared__ float red[256];
  int b = threadIdx.x;
  float s = 0.f;
  for (int zc = 0; zc < Z_; ++zc) {
    size_t idx = (size_t)b * T_ * Z_ + zc;  // t = 0
    float mq = mu_q[idx], lq = lv_q[idx];
    s += 0.5f * (-lq + expf(lq) + mq * mq - 1.f);
  }
  red[b] = fmaxf(s, 0.5f);
  __syncthreads();
  if (b == 0) {
    float tot = 0.f;
    for (int i = 0; i < 256; ++i) tot += red[i];
    atomicAdd(out + 1, tot * (1.f / (float)B_));
  }
}

// ---------------------------------------------------------------------------
extern "C" void kernel_launch(void* const* d_in, const int* in_sizes, int n_in,
                              void* d_out, int out_size, void* d_ws, size_t ws_size,
                              hipStream_t stream) {
  (void)in_sizes; (void)n_in; (void)out_size; (void)ws_size;
  const float* y    = (const float*)d_in[0];
  const float* u    = (const float*)d_in[1];
  const float* eps  = (const float*)d_in[2];
  const float* w_ih = (const float*)d_in[3];
  const float* w_hh = (const float*)d_in[4];
  const float* b_ih = (const float*)d_in[5];
  const float* b_hh = (const float*)d_in[6];
  const float* wq   = (const float*)d_in[7];
  const float* bq   = (const float*)d_in[8];
  const float* tw1  = (const float*)d_in[9];
  const float* tb1  = (const float*)d_in[10];
  const float* tw2  = (const float*)d_in[11];
  const float* tb2  = (const float*)d_in[12];
  const float* tw3  = (const float*)d_in[13];
  const float* tb3  = (const float*)d_in[14];
  const float* ew1  = (const float*)d_in[15];
  const float* eb1  = (const float*)d_in[16];
  const float* ew2  = (const float*)d_in[17];
  const float* eb2  = (const float*)d_in[18];
  const float* ew3  = (const float*)d_in[19];
  const float* eb3  = (const float*)d_in[20];
  const float* lsy  = (const float*)d_in[21];
  float* out = (float*)d_out;

  // workspace carve-up (256B aligned)
  char* p = (char*)d_ws;
  auto carve = [&](size_t bytes) -> void* {
    void* r = (void*)p;
    p += (bytes + 255) & ~(size_t)255;
    return r;
  };
  _Float16* wih_h = (_Float16*)carve((size_t)768 * 64 * 2);
  _Float16* whh_h = (_Float16*)carve((size_t)768 * 256 * 2);
  _Float16* wq_h  = (_Float16*)carve((size_t)128 * 256 * 2);
  _Float16* tw1_h = (_Float16*)carve((size_t)512 * 96 * 2);
  _Float16* tw2_h = (_Float16*)carve((size_t)512 * 512 * 2);
  _Float16* tw3_h = (_Float16*)carve((size_t)128 * 512 * 2);
  _Float16* ew1_h = (_Float16*)carve((size_t)512 * 64 * 2);
  _Float16* ew2_h = (_Float16*)carve((size_t)512 * 512 * 2);
  _Float16* ew3_h = (_Float16*)carve((size_t)32 * 512 * 2);
  float*    mu_q  = (float*)carve((size_t)B_ * T_ * Z_ * 4);
  float*    lv_q  = (float*)carve((size_t)B_ * T_ * Z_ * 4);
  _Float16* z_h   = (_Float16*)carve((size_t)B_ * T_ * Z_ * 2);

  zero2_kernel<<<1, 1, 0, stream>>>(out);

  auto cvt = [&](const float* s, _Float16* d, int n) {
    cvt_f32_f16<<<(n + 255) / 256, 256, 0, stream>>>(s, d, n);
  };
  cvt_pad_f16<<<(768 * 64 + 255) / 256, 256, 0, stream>>>(w_ih, wih_h, 768, 48, 64);
  cvt(w_hh, whh_h, 768 * 256);
  cvt(wq, wq_h, 128 * 256);
  cvt_pad_f16<<<(512 * 96 + 255) / 256, 256, 0, stream>>>(tw1, tw1_h, 512, 80, 96);
  cvt(tw2, tw2_h, 512 * 512);
  cvt(tw3, tw3_h, 128 * 512);
  cvt(ew1, ew1_h, 512 * 64);
  cvt(ew2, ew2_h, 512 * 512);
  cvt(ew3, ew3_h, 32 * 512);

  gru_scan_kernel<<<B_ / 16, 256, 0, stream>>>(y, u, eps, wih_h, whh_h, wq_h,
                                               b_ih, b_hh, bq, mu_q, lv_q, z_h);
  emission_nll_kernel<<<(B_ * T_) / 16, 256, 0, stream>>>(
      z_h, y, ew1_h, eb1, ew2_h, eb2, ew3_h, eb3, lsy, out);
  trans_kl_kernel<<<B_ * 32, 256, 0, stream>>>(z_h, u, mu_q, lv_q, tw1_h, tb1,
                                               tw2_h, tb2, tw3_h, tb3, out);
  kl0_kernel<<<1, 256, 0, stream>>>(mu_q, lv_q, out);
}
